// FlashPhi3SmallLayer_23545010717555
// MI455X (gfx1250) — matrix operand order repaired and checked by reading, not compiled, not run on hardware
//
#include <hip/hip_runtime.h>
#include <hip/hip_bf16.h>

// ---------------------------------------------------------------------------
// Shapes (compile-time; match reference)
// ---------------------------------------------------------------------------
constexpr int Tn    = 4096;     // B*S tokens
constexpr int Bn    = 4;
constexpr int Sn    = 1024;
constexpr int HIDn  = 4096;
constexpr int NHn   = 32;
constexpr int NKVn  = 8;
constexpr int HDn   = 128;
constexpr int QKVN  = (NHn + 2 * NKVn) * HDn;   // 6144
constexpr int INTERn = 14336;
constexpr int UPN    = 2 * INTERn;              // 28672
constexpr float LIMITf = 20.0f;
constexpr float EPSf   = 1e-5f;
constexpr float SCALEf = 0.08838834764831845f;  // 128^-0.5

typedef __bf16 bf16;
typedef __attribute__((ext_vector_type(16))) __bf16 v16bf;
typedef __attribute__((ext_vector_type(8)))  __bf16 v8bf;
typedef __attribute__((ext_vector_type(8)))  float  v8f;

#if __has_builtin(__builtin_amdgcn_tensor_load_to_lds) && \
    __has_builtin(__builtin_amdgcn_s_wait_tensorcnt)
#define USE_TDM 1
#endif

__device__ inline v8f vzero8f() {
    v8f z;
#pragma unroll
    for (int i = 0; i < 8; ++i) z[i] = 0.0f;
    return z;
}

__device__ inline v8f wmma_bf16(v16bf a, v16bf b, v8f c) {
    // D = A(16x32 bf16) * B(32x16 bf16) + C(16x16 f32)
    return __builtin_amdgcn_wmma_f32_16x16x32_bf16(
        /*neg_a=*/false, a, /*neg_b=*/false, b,
        /*c_mod=*/(short)0, c, /*reuse_a=*/false, /*reuse_b=*/false);
}

// A-fragment (16x32, 16-bit): lane l holds row M=l%16, 16 consecutive K
// starting at (l/16)*16 (matches B fragment). Two 16B LDS loads.
template <int LD>
__device__ inline v16bf load_a_frag(const bf16* base, int lane) {
    const int row  = lane & 15;
    const int koff = (lane >> 4) * 16;
    const bf16* p = base + row * LD + koff;
    v8bf lo = *(const v8bf*)(p);
    v8bf hi = *(const v8bf*)(p + 8);
    v16bf r;
#pragma unroll
    for (int i = 0; i < 8; ++i) { r[i] = lo[i]; r[i + 8] = hi[i]; }
    return r;
}

// B-fragment (32x16, 16-bit): lane l holds col N=l%16, 16 consecutive K
// starting at (l/16)*16. Requires B stored [N][K] in LDS -> one 32B load.
template <int LD>
__device__ inline v16bf load_b_frag(const bf16* base, int lane) {
    const int n    = lane & 15;
    const int koff = (lane >> 4) * 16;
    return *(const v16bf*)(base + n * LD + koff);
}

__device__ inline float wave_sum32(float v) {
#pragma unroll
    for (int off = 16; off; off >>= 1) v += __shfl_xor(v, off);
    return v;
}

#ifdef USE_TDM
// ---------------------------------------------------------------------------
// Tensor Data Mover: async DMA of a 2D tile (rows x tile_k bf16, row stride
// strideElems) from global into LDS.  D# per CDNA5 ISA section 8.3/8.4:
//   group0: [1:0]count=1 | lds_addr[63:32] | global_addr[120:64] | type=2
//   group1: wg_mask=0, data_size=1(2B), tensor_dim0/1, tile_dim0/1, stride0
//   groups 2/3: zero (tile_dim3/4 unused -> 2D)
// Tracked by TENSORcnt; issued by one wave per workgroup.
// ---------------------------------------------------------------------------
typedef __attribute__((ext_vector_type(4))) unsigned int u32x4;
typedef __attribute__((ext_vector_type(8))) int i32x8;
typedef __attribute__((ext_vector_type(4))) int i32x4;

__device__ inline void tdm_load_2d(const void* gaddr, void* ldsaddr,
                                   unsigned tile_k, unsigned rows,
                                   unsigned strideElems) {
    unsigned long long ga = (unsigned long long)(size_t)gaddr;
    unsigned lo = (unsigned)(size_t)ldsaddr;   // LDS aperture: low 32 bits = offset
    u32x4 g0;
    g0[0] = 1u;                                        // count=1 (valid user D#)
    g0[1] = lo;                                        // lds_addr
    g0[2] = (unsigned)(ga & 0xFFFFFFFFu);              // global_addr[31:0]
    g0[3] = (unsigned)((ga >> 32) & 0x01FFFFFFu)       // global_addr[56:32]
            | (2u << 30);                              // type=2 ("image")
    i32x8 g1;
    g1[0] = (int)(1u << 16);                           // data_size=1 -> 2 bytes
    g1[1] = (int)(tile_k << 16);                       // tensor_dim0[15:0]
    g1[2] = (int)((tile_k >> 16) & 0xFFFFu)            // tensor_dim0[31:16]
            | (int)(rows << 16);                       // tensor_dim1[15:0]
    g1[3] = (int)(tile_k << 16);                       // tensor_dim1[31:16]=0 | tile_dim0
    g1[4] = (int)(rows & 0xFFFFu);                     // tile_dim1 | tile_dim2=0
    g1[5] = (int)strideElems;                          // tensor_dim0_stride[31:0]
    g1[6] = 0;                                         // stride0[47:32] | stride1 lo
    g1[7] = 0;
    i32x4 z4 = {0, 0, 0, 0};
#if defined(__clang_major__) && __clang_major__ >= 23
    i32x8 z8 = {0, 0, 0, 0, 0, 0, 0, 0};
    __builtin_amdgcn_tensor_load_to_lds(g0, g1, z4, z4, z8, 0);
#else
    __builtin_amdgcn_tensor_load_to_lds(g0, g1, z4, z4, 0);
#endif
}
#endif  // USE_TDM

// ---------------------------------------------------------------------------
// 1) fp32 (K x N) -> bf16 (N x K) tiled transpose-convert
// ---------------------------------------------------------------------------
__global__ void conv_tr_kernel(const float* __restrict__ src, bf16* __restrict__ dst,
                               int K, int N) {
    __shared__ float tile[32][33];
    const int tx = threadIdx.x, ty = threadIdx.y;   // block (32,8)
    const int bx = blockIdx.x, by = blockIdx.y;
#pragma unroll
    for (int i = 0; i < 4; ++i) {
        int kk = by * 32 + ty + 8 * i;
        tile[ty + 8 * i][tx] = src[(size_t)kk * N + bx * 32 + tx];
    }
    __syncthreads();
#pragma unroll
    for (int i = 0; i < 4; ++i) {
        int n = bx * 32 + ty + 8 * i;
        int k = by * 32 + tx;
        dst[(size_t)n * K + k] = (bf16)tile[tx][ty + 8 * i];
    }
}

// ---------------------------------------------------------------------------
// 2) fused residual-add + LayerNorm: hout=f32(x+res), normed=bf16(LN(hout))
// ---------------------------------------------------------------------------
__global__ __launch_bounds__(256) void add_ln_kernel(
    const float* __restrict__ x, const float* __restrict__ res,
    const float* __restrict__ w, const float* __restrict__ bb,
    float* __restrict__ hout, bf16* __restrict__ normed) {
    const int t   = blockIdx.x;
    const int tid = threadIdx.x;
    const int lane = tid & 31, wave = tid >> 5;
    const float4* X = (const float4*)(x   + (size_t)t * HIDn);
    const float4* R = (const float4*)(res + (size_t)t * HIDn);
    float4* H = (float4*)(hout + (size_t)t * HIDn);
    bf16*   O = normed + (size_t)t * HIDn;

    float4 v4[4];
    float s = 0.f, s2 = 0.f;
#pragma unroll
    for (int i = 0; i < 4; ++i) {
        int c = tid + 256 * i;
        float4 a = X[c], b = R[c];
        float4 v; v.x = a.x + b.x; v.y = a.y + b.y; v.z = a.z + b.z; v.w = a.w + b.w;
        v4[i] = v;
        s  += v.x + v.y + v.z + v.w;
        s2 += v.x * v.x + v.y * v.y + v.z * v.z + v.w * v.w;
    }
    s = wave_sum32(s); s2 = wave_sum32(s2);
    __shared__ float red[2][8];
    if (lane == 0) { red[0][wave] = s; red[1][wave] = s2; }
    __syncthreads();
    float ts = 0.f, ts2 = 0.f;
#pragma unroll
    for (int wv = 0; wv < 8; ++wv) { ts += red[0][wv]; ts2 += red[1][wv]; }
    const float mean = ts * (1.0f / HIDn);
    const float var  = ts2 * (1.0f / HIDn) - mean * mean;
    const float rstd = rsqrtf(var + EPSf);
#pragma unroll
    for (int i = 0; i < 4; ++i) {
        int c = tid + 256 * i;
        H[c] = v4[i];
        const float vv[4] = { v4[i].x, v4[i].y, v4[i].z, v4[i].w };
#pragma unroll
        for (int j = 0; j < 4; ++j) {
            int col = c * 4 + j;
            O[col] = (bf16)((vv[j] - mean) * rstd * w[col] + bb[col]);
        }
    }
}

// ---------------------------------------------------------------------------
// 3) WMMA GEMM: C(MxN) = A(MxK,bf16) * Bt(NxK,bf16)^T + bias
//    BM=BN=128, BK=32; 256 thr = 8 waves in 4x2; per wave 32x64 (2x4 tiles)
//    EPI 0: f32 store (+bias).  EPI 1: fused gegelu -> bf16, N halves.
//    Tile movement: TDM double-buffered (wave 0 issues descriptors, all
//    waves compute), falling back to register-staged VMEM if TDM missing.
// ---------------------------------------------------------------------------
constexpr int BM = 128, BN = 128, BK = 32;

template <int EPI>
__global__ __launch_bounds__(256, 2) void gemm_bf16_wmma(
    const bf16* __restrict__ A, const bf16* __restrict__ Bt,
    const float* __restrict__ bias, void* __restrict__ Cout,
    int M, int N, int K) {
    const int tid  = threadIdx.x;
    const int lane = tid & 31;
    const int wave = tid >> 5;
    const int wm = wave >> 1;   // 0..3 -> 32 rows each
    const int wn = wave & 1;    // 0..1 -> 64 cols each
    const int bm = blockIdx.y * BM;
    const int bn = blockIdx.x * BN;
    const int nk = K / BK;

    v8f acc[2][4];
#pragma unroll
    for (int mi = 0; mi < 2; ++mi)
#pragma unroll
        for (int nt = 0; nt < 4; ++nt) acc[mi][nt] = vzero8f();

#ifdef USE_TDM
    __shared__ bf16 As[2][BM][BK];
    __shared__ bf16 Bs[2][BN][BK];
    const bf16* Abase = A  + (size_t)bm * K;
    const bf16* Bbase = Bt + (size_t)bn * K;

    if (wave == 0) {
        tdm_load_2d(Abase, &As[0][0][0], BK, BM, (unsigned)K);
        tdm_load_2d(Bbase, &Bs[0][0][0], BK, BN, (unsigned)K);
        __builtin_amdgcn_s_wait_tensorcnt(0);
    }
    __syncthreads();

    for (int kt = 0; kt < nk; ++kt) {
        const int buf = kt & 1;
        if (wave == 0 && kt + 1 < nk) {
            tdm_load_2d(Abase + (size_t)(kt + 1) * BK, &As[buf ^ 1][0][0], BK, BM, (unsigned)K);
            tdm_load_2d(Bbase + (size_t)(kt + 1) * BK, &Bs[buf ^ 1][0][0], BK, BN, (unsigned)K);
        }
        v16bf a0 = load_a_frag<BK>(&As[buf][wm * 32][0], lane);
        v16bf a1 = load_a_frag<BK>(&As[buf][wm * 32 + 16][0], lane);
#pragma unroll
        for (int nt = 0; nt < 4; ++nt) {
            v16bf b = load_b_frag<BK>(&Bs[buf][wn * 64 + nt * 16][0], lane);
            acc[0][nt] = wmma_bf16(a0, b, acc[0][nt]);
            acc[1][nt] = wmma_bf16(a1, b, acc[1][nt]);
        }
        if (kt + 1 < nk) {
            if (wave == 0) __builtin_amdgcn_s_wait_tensorcnt(0);
            __syncthreads();
        }
    }
#else
    __shared__ bf16 As[BM][BK];
    __shared__ bf16 Bs[BN][BK];
    // 128x32 bf16 tile = 512 x 16B chunks; 2 chunks per thread
    auto fetchA = [&](int kt, uint4* r) {
#pragma unroll
        for (int i = 0; i < 2; ++i) {
            int c = tid + 256 * i, row = c >> 2, col = (c & 3) * 8;
            r[i] = *(const uint4*)&A[(size_t)(bm + row) * K + (size_t)kt * BK + col];
        }
    };
    auto fetchB = [&](int kt, uint4* r) {
#pragma unroll
        for (int i = 0; i < 2; ++i) {
            int c = tid + 256 * i, row = c >> 2, col = (c & 3) * 8;
            r[i] = *(const uint4*)&Bt[(size_t)(bn + row) * K + (size_t)kt * BK + col];
        }
    };
    auto stA = [&](const uint4* r) {
#pragma unroll
        for (int i = 0; i < 2; ++i) {
            int c = tid + 256 * i, row = c >> 2, col = (c & 3) * 8;
            *(uint4*)&As[row][col] = r[i];
        }
    };
    auto stB = [&](const uint4* r) {
#pragma unroll
        for (int i = 0; i < 2; ++i) {
            int c = tid + 256 * i, row = c >> 2, col = (c & 3) * 8;
            *(uint4*)&Bs[row][col] = r[i];
        }
    };
    uint4 ra[2], rb[2], na[2], nb[2];
    fetchA(0, ra); fetchB(0, rb);
    stA(ra); stB(rb);
    __syncthreads();
    for (int kt = 0; kt < nk; ++kt) {
        if (kt + 1 < nk) { fetchA(kt + 1, na); fetchB(kt + 1, nb); }
        v16bf a0 = load_a_frag<BK>(&As[wm * 32][0], lane);
        v16bf a1 = load_a_frag<BK>(&As[wm * 32 + 16][0], lane);
#pragma unroll
        for (int nt = 0; nt < 4; ++nt) {
            v16bf b = load_b_frag<BK>(&Bs[wn * 64 + nt * 16][0], lane);
            acc[0][nt] = wmma_bf16(a0, b, acc[0][nt]);
            acc[1][nt] = wmma_bf16(a1, b, acc[1][nt]);
        }
        __syncthreads();
        if (kt + 1 < nk) { stA(na); stB(nb); }
        __syncthreads();
    }
#endif

    // Epilogue. C layout: lane l, vgpr g -> row = g + 8*(l/16), col = l%16
    const int hw = lane >> 4, ln = lane & 15;
#pragma unroll
    for (int mi = 0; mi < 2; ++mi)
#pragma unroll
        for (int nt = 0; nt < 4; ++nt)
#pragma unroll
            for (int g = 0; g < 8; ++g) {
                int row = bm + wm * 32 + mi * 16 + g + 8 * hw;
                int col = bn + wn * 64 + nt * 16 + ln;
                float vv = acc[mi][nt][g];
                if (bias) vv += bias[col];
                if (EPI == 0) {
                    ((float*)Cout)[(size_t)row * N + col] = vv;
                } else {
                    // gegelu: even col = gate input, odd col = linear input
                    float other = __shfl_xor(vv, 1);
                    if ((ln & 1) == 0) {
                        float a   = fminf(vv, LIMITf);
                        float lin = fminf(fmaxf(other, -LIMITf), LIMITf);
                        float r = a * (1.0f / (1.0f + __expf(-1.702f * a))) * (lin + 1.0f);
                        ((bf16*)Cout)[(size_t)row * (N >> 1) + (col >> 1)] = (bf16)r;
                    }
                }
            }
}

// ---------------------------------------------------------------------------
// 4) RoPE + split: qkv(f32, T x 6144) -> q/k/v bf16
// ---------------------------------------------------------------------------
__global__ __launch_bounds__(256) void rope_kernel(
    const float* __restrict__ qkv, const float* __restrict__ cosp,
    const float* __restrict__ sinp, bf16* __restrict__ qo,
    bf16* __restrict__ ko, bf16* __restrict__ vo) {
    const int t = blockIdx.x;
    const float* row = qkv + (size_t)t * QKVN;
    for (int idx = threadIdx.x; idx < NHn * 64; idx += blockDim.x) {
        int hh = idx >> 6, d = idx & 63;
        float x1 = row[hh * HDn + d], x2 = row[hh * HDn + 64 + d];
        float c = cosp[(size_t)t * 64 + d], s = sinp[(size_t)t * 64 + d];
        qo[((size_t)t * NHn + hh) * HDn + d]      = (bf16)(x1 * c - x2 * s);
        qo[((size_t)t * NHn + hh) * HDn + 64 + d] = (bf16)(x2 * c + x1 * s);
    }
    const float* krow = row + NHn * HDn;
    for (int idx = threadIdx.x; idx < NKVn * 64; idx += blockDim.x) {
        int hh = idx >> 6, d = idx & 63;
        float x1 = krow[hh * HDn + d], x2 = krow[hh * HDn + 64 + d];
        float c = cosp[(size_t)t * 64 + d], s = sinp[(size_t)t * 64 + d];
        ko[((size_t)t * NKVn + hh) * HDn + d]      = (bf16)(x1 * c - x2 * s);
        ko[((size_t)t * NKVn + hh) * HDn + 64 + d] = (bf16)(x2 * c + x1 * s);
    }
    const float* vrow = krow + NKVn * HDn;
    for (int idx = threadIdx.x; idx < NKVn * HDn; idx += blockDim.x)
        vo[(size_t)t * NKVn * HDn + idx] = (bf16)vrow[idx];
}

// ---------------------------------------------------------------------------
// 5) Flash attention (causal, GQA). Block = 128 thr (4 waves); each block
//    handles one (b,h) and a 64-row Q tile; 64-col K/V tiles, online softmax.
// ---------------------------------------------------------------------------
__global__ __launch_bounds__(128, 2) void attn_kernel(
    const bf16* __restrict__ q, const bf16* __restrict__ k,
    const bf16* __restrict__ v, bf16* __restrict__ ctx) {
    __shared__ bf16 Qs[64][HDn];     // 16 KB
    __shared__ bf16 Ks[64][HDn];     // 16 KB   [kv_row][d]
    __shared__ bf16 Vt[HDn][64];     // 16 KB   [d][kv_row] (transposed)
    __shared__ bf16 Ps[4][16][64];   //  8 KB   per-wave probabilities

    const int bh  = blockIdx.x;      // b*NH + h
    const int b   = bh >> 5;
    const int h   = bh & 31;
    const int kvh = h >> 2;          // NH/NKV = 4
    const int qt  = blockIdx.y;      // 0..15
    const int tid = threadIdx.x;
    const int lane = tid & 31, wave = tid >> 5;
    const int hw = lane >> 4, ln = lane & 15;

    // load Q tile (64 x 128): 1024 x 16B chunks / 128 thr = 8 each
#pragma unroll
    for (int i = 0; i < 8; ++i) {
        int c = tid + 128 * i, row = c >> 4, col = (c & 15) * 8;
        int tok = b * Sn + qt * 64 + row;
        *(uint4*)&Qs[row][col] = *(const uint4*)&q[((size_t)tok * NHn + h) * HDn + col];
    }

    v8f oacc[8];
#pragma unroll
    for (int d = 0; d < 8; ++d) oacc[d] = vzero8f();
    float mrow[8], lrow[8];
#pragma unroll
    for (int g = 0; g < 8; ++g) { mrow[g] = -3.0e38f; lrow[g] = 0.0f; }

    for (int kt = 0; kt <= qt; ++kt) {
        __syncthreads();
        // K tile, natural layout
#pragma unroll
        for (int i = 0; i < 8; ++i) {
            int c = tid + 128 * i, row = c >> 4, col = (c & 15) * 8;
            int tok = b * Sn + kt * 64 + row;
            *(uint4*)&Ks[row][col] =
                *(const uint4*)&k[((size_t)tok * NKVn + kvh) * HDn + col];
        }
        // V tile transposed: two k-rows per thread, pack bf16 pairs -> b32 LDS
#pragma unroll
        for (int i = 0; i < 4; ++i) {
            int c = tid + 128 * i;            // 0..511
            int rp = c >> 4;                  // row pair 0..31
            int col = (c & 15) * 8;
            int tok0 = b * Sn + kt * 64 + rp * 2;
            uint4 r0 = *(const uint4*)&v[((size_t)tok0 * NKVn + kvh) * HDn + col];
            uint4 r1 = *(const uint4*)&v[((size_t)(tok0 + 1) * NKVn + kvh) * HDn + col];
            const unsigned short* a0 = (const unsigned short*)&r0;
            const unsigned short* a1 = (const unsigned short*)&r1;
#pragma unroll
            for (int j = 0; j < 8; ++j) {
                unsigned int pk = (unsigned)a0[j] | ((unsigned)a1[j] << 16);
                *(unsigned int*)&Vt[col + j][rp * 2] = pk;
            }
        }
        __syncthreads();

        // S = Q * K^T  (per wave: 16 rows x 64 cols, K-dim = HD = 4x32)
        v8f sc[4];
#pragma unroll
        for (int nt = 0; nt < 4; ++nt) sc[nt] = vzero8f();
#pragma unroll
        for (int kk = 0; kk < 4; ++kk) {
            v16bf aq = load_a_frag<HDn>(&Qs[wave * 16][kk * 32], lane);
#pragma unroll
            for (int nt = 0; nt < 4; ++nt) {
                v16bf bk = load_b_frag<HDn>(&Ks[nt * 16][kk * 32], lane);
                sc[nt] = wmma_bf16(aq, bk, sc[nt]);
            }
        }

        // scale + causal mask + online softmax (row stats over half-wave)
#pragma unroll
        for (int g = 0; g < 8; ++g) {
            const int row_g = qt * 64 + wave * 16 + g + 8 * hw;
            float mx = -3.0e38f;
#pragma unroll
            for (int nt = 0; nt < 4; ++nt) {
                float val = sc[nt][g] * SCALEf;
                int colg = kt * 64 + nt * 16 + ln;
                if (colg > row_g) val = -3.0e38f;
                sc[nt][g] = val;
                mx = fmaxf(mx, val);
            }
#pragma unroll
            for (int off = 8; off; off >>= 1) mx = fmaxf(mx, __shfl_xor(mx, off));
            const float mnew  = fmaxf(mrow[g], mx);
            const float alpha = __expf(mrow[g] - mnew);
            float psum = 0.0f;
#pragma unroll
            for (int nt = 0; nt < 4; ++nt) {
                float p = __expf(sc[nt][g] - mnew);
                sc[nt][g] = p;
                psum += p;
            }
#pragma unroll
            for (int off = 8; off; off >>= 1) psum += __shfl_xor(psum, off);
            lrow[g] = lrow[g] * alpha + psum;
            mrow[g] = mnew;
#pragma unroll
            for (int d = 0; d < 8; ++d) oacc[d][g] *= alpha;
#pragma unroll
            for (int nt = 0; nt < 4; ++nt)
                Ps[wave][g + 8 * hw][nt * 16 + ln] = (bf16)sc[nt][g];
        }

        // O += P * V   (K-dim = 64 = 2x32); same-wave LDS RAW, HW waits handle it
#pragma unroll
        for (int kk = 0; kk < 2; ++kk) {
            v16bf ap = load_a_frag<64>(&Ps[wave][0][kk * 32], lane);
#pragma unroll
            for (int dt = 0; dt < 8; ++dt) {
                v16bf bv = load_b_frag<64>(&Vt[dt * 16][kk * 32], lane);
                oacc[dt] = wmma_bf16(ap, bv, oacc[dt]);
            }
        }
    }

    // normalize + store ctx (T, NH*HD) bf16
#pragma unroll
    for (int dt = 0; dt < 8; ++dt)
#pragma unroll
        for (int g = 0; g < 8; ++g) {
            int row = wave * 16 + g + 8 * hw;
            int tok = b * Sn + qt * 64 + row;
            ctx[((size_t)tok * NHn + h) * HDn + dt * 16 + ln] =
                (bf16)(oacc[dt][g] / lrow[g]);
        }
}

// ---------------------------------------------------------------------------
// Host launcher
// ---------------------------------------------------------------------------
extern "C" void kernel_launch(void* const* d_in, const int* in_sizes, int n_in,
                              void* d_out, int out_size, void* d_ws, size_t ws_size,
                              hipStream_t stream) {
    (void)in_sizes; (void)n_in; (void)out_size; (void)ws_size;
    const float* hidden   = (const float*)d_in[0];
    const float* residual = (const float*)d_in[1];
    const float* cosp     = (const float*)d_in[2];
    const float* sinp     = (const float*)d_in[3];
    /* d_in[4] = cu_seqlen_prefill (unused; fixed S) */
    const float* w_qkv  = (const float*)d_in[5];
    const float* b_qkv  = (const float*)d_in[6];
    const float* w_o    = (const float*)d_in[7];
    const float* w_ln1  = (const float*)d_in[8];
    const float* b_ln1  = (const float*)d_in[9];
    const float* w_ln2  = (const float*)d_in[10];
    const float* b_ln2  = (const float*)d_in[11];
    const float* w_up   = (const float*)d_in[12];
    const float* b_up   = (const float*)d_in[13];
    const float* w_down = (const float*)d_in[14];
    const float* b_down = (const float*)d_in[15];

    char* ws = (char*)d_ws;
    size_t off = 0;
    auto take = [&](size_t bytes) -> char* {
        char* p = ws + off;
        off += (bytes + 255) & ~(size_t)255;
        return p;
    };
    bf16* wq_b  = (bf16*)take((size_t)HIDn * QKVN * 2);
    bf16* wo_b  = (bf16*)take((size_t)HIDn * HIDn * 2);
    bf16* wu_b  = (bf16*)take((size_t)HIDn * UPN * 2);
    bf16* wd_b  = (bf16*)take((size_t)INTERn * HIDn * 2);
    float* h_f      = (float*)take((size_t)Tn * HIDn * 4);
    bf16*  normed_b = (bf16*)take((size_t)Tn * HIDn * 2);
    float* qkv_f    = (float*)take((size_t)Tn * QKVN * 4);
    bf16*  q_b      = (bf16*)take((size_t)Tn * NHn * HDn * 2);
    bf16*  k_b      = (bf16*)take((size_t)Tn * NKVn * HDn * 2);
    bf16*  v_b      = (bf16*)take((size_t)Tn * NKVn * HDn * 2);
    bf16*  ctx_b    = (bf16*)take((size_t)Tn * HIDn * 2);
    float* attn_f   = (float*)take((size_t)Tn * HIDn * 4);
    bf16*  normed2_b = (bf16*)take((size_t)Tn * HIDn * 2);
    bf16*  act_b    = (bf16*)take((size_t)Tn * INTERn * 2);

    float* mlp_out = (float*)d_out;
    float* h2_out  = (float*)d_out + (size_t)Tn * HIDn;

    const dim3 tb(32, 8);
    conv_tr_kernel<<<dim3(QKVN / 32, HIDn / 32), tb, 0, stream>>>(w_qkv, wq_b, HIDn, QKVN);
    conv_tr_kernel<<<dim3(HIDn / 32, HIDn / 32), tb, 0, stream>>>(w_o, wo_b, HIDn, HIDn);
    conv_tr_kernel<<<dim3(UPN / 32, HIDn / 32), tb, 0, stream>>>(w_up, wu_b, HIDn, UPN);
    conv_tr_kernel<<<dim3(HIDn / 32, INTERn / 32), tb, 0, stream>>>(w_down, wd_b, INTERn, HIDn);

    add_ln_kernel<<<Tn, 256, 0, stream>>>(hidden, residual, w_ln1, b_ln1, h_f, normed_b);

    gemm_bf16_wmma<0><<<dim3(QKVN / BN, Tn / BM), 256, 0, stream>>>(
        normed_b, wq_b, b_qkv, qkv_f, Tn, QKVN, HIDn);

    rope_kernel<<<Tn, 256, 0, stream>>>(qkv_f, cosp, sinp, q_b, k_b, v_b);

    attn_kernel<<<dim3(Bn * NHn, Sn / 64), 128, 0, stream>>>(q_b, k_b, v_b, ctx_b);

    gemm_bf16_wmma<0><<<dim3(HIDn / BN, Tn / BM), 256, 0, stream>>>(
        ctx_b, wo_b, nullptr, attn_f, Tn, HIDn, HIDn);

    add_ln_kernel<<<Tn, 256, 0, stream>>>(attn_f, h_f, w_ln2, b_ln2, h2_out, normed2_b);

    gemm_bf16_wmma<1><<<dim3(UPN / BN, Tn / BM), 256, 0, stream>>>(
        normed2_b, wu_b, b_up, act_b, Tn, UPN, HIDn);

    gemm_bf16_wmma<0><<<dim3(HIDn / BN, Tn / BM), 256, 0, stream>>>(
        act_b, wd_b, b_down, mlp_out, Tn, HIDn, INTERn);
}
